// WanSelfAttention_71957882077554
// MI455X (gfx1250) — matrix operand order repaired and checked by reading, not compiled, not run on hardware
//
#include <hip/hip_runtime.h>
#include <hip/hip_bf16.h>

// ---------------- problem constants (match reference setup) ----------------
#define S_LEN 3840      // 6*20*32
#define DIM   1536
#define NH    12
#define HD    128
#define CPLX  64
#define QKVN  4608      // 3*DIM
#define F_G   6
#define H_G   20
#define W_G   32

typedef __attribute__((ext_vector_type(16))) _Float16 v16h;
typedef __attribute__((ext_vector_type(8)))  _Float16 v8h;
typedef __attribute__((ext_vector_type(8)))  float    v8f;

// ---------------- WMMA fragment helpers (gfx1250 wave32 layouts) ----------------
// A fragment, 16x32 f16, row-major source with leading dim ld.
// ISA layout: lanes 0-15 row M=lane, halves = K {0..7, 16..23};
//             lanes 16-31 row M=lane-16, halves = K {8..15, 24..31}.
__device__ __forceinline__ v16h load_a_frag(const _Float16* base, int ld, int m0, int k0) {
    int lane = threadIdx.x & 31;
    const _Float16* row = base + (size_t)(m0 + (lane & 15)) * ld + k0 + ((lane >> 4) << 3);
    v8h lo = *(const v8h*)(row);
    v8h hi = *(const v8h*)(row + 16);
    return __builtin_shufflevector(lo, hi, 0,1,2,3,4,5,6,7,8,9,10,11,12,13,14,15);
}

// B fragment, 32x16 f16 where B[k][n] = src[n][k] (src row-major, leading dim ld).
// ISA layout: lanes 0-15 col N=lane, halves = K {0..7, 16..23}; lanes 16-31 col N=lane-16, K {8..15, 24..31}.
__device__ __forceinline__ v16h load_b_frag(const _Float16* base, int ld, int n0, int k0) {
    int lane = threadIdx.x & 31;
    const _Float16* p = base + (size_t)(n0 + (lane & 15)) * ld + k0 + ((lane >> 4) << 3);
    v8h lo = *(const v8h*)(p);
    v8h hi = *(const v8h*)(p + 16);
    return __builtin_shufflevector(lo, hi, 0,1,2,3,4,5,6,7,8,9,10,11,12,13,14,15);
}

__device__ __forceinline__ v8f wmma_f16(v16h a, v16h b, v8f c) {
    return __builtin_amdgcn_wmma_f32_16x16x32_f16(false, a, false, b, (short)0, c, false, false);
}

// ---------------- kernel 0: precision conversion / packing ----------------
__global__ void prep_kernel(const float* __restrict__ x,
                            const float* __restrict__ Wq, const float* __restrict__ Wk,
                            const float* __restrict__ Wv, const float* __restrict__ Wo,
                            const float* __restrict__ bq, const float* __restrict__ bk,
                            const float* __restrict__ bv,
                            _Float16* __restrict__ xh, _Float16* __restrict__ wqkvh,
                            _Float16* __restrict__ woh, float* __restrict__ bqkv) {
    const size_t NX  = (size_t)S_LEN * DIM;     // 5,898,240
    const size_t NW  = (size_t)QKVN * DIM;      // 7,077,888 (largest)
    const size_t NWO = (size_t)DIM * DIM;
    for (size_t i = (size_t)blockIdx.x * blockDim.x + threadIdx.x; i < NW;
         i += (size_t)gridDim.x * blockDim.x) {
        if (i < NX)  xh[i]  = (_Float16)x[i];
        {
            int n = (int)(i / DIM), kk = (int)(i % DIM);
            const float* Wsrc = (n < DIM) ? Wq : ((n < 2 * DIM) ? Wk : Wv);
            int nn = n % DIM;
            wqkvh[i] = (_Float16)Wsrc[(size_t)nn * DIM + kk];
        }
        if (i < NWO) woh[i] = (_Float16)Wo[i];
        if (i < QKVN) bqkv[i] = (i < DIM) ? bq[i] : (i < 2 * DIM) ? bk[i - DIM] : bv[i - 2 * DIM];
    }
}

// ---------------- kernels 1 & 4: WMMA GEMM  C[M][N] = A[M][K] @ W[N][K]^T + bias ----------------
// One wave computes a 16x64 output tile. Two alternating fragment buffers are
// reloaded in place after their WMMAs consume them (in-order issue resolves the
// WAR), so VMEM latency hides behind the other buffer's matrix ops with no
// register-rotation copies. K must be a multiple of 64 (1536 is).
__global__ void gemm_wmma(const _Float16* __restrict__ A, const _Float16* __restrict__ W,
                          const float* __restrict__ bias, float* __restrict__ C,
                          int M, int N, int K, int mtiles, int ntiles) {
    int wave = (int)((blockIdx.x * blockDim.x + threadIdx.x) >> 5);
    if (wave >= mtiles * ntiles) return;
    int mt = wave % mtiles, nt = wave / mtiles;
    int m0 = mt * 16, n0 = nt * 64;

    v8f acc[4] = {};
    v16h a0, a1, b0[4], b1[4];
    a0 = load_a_frag(A, K, m0, 0);
#pragma unroll
    for (int j = 0; j < 4; ++j) b0[j] = load_b_frag(W, K, n0 + j * 16, 0);
    a1 = load_a_frag(A, K, m0, 32);
#pragma unroll
    for (int j = 0; j < 4; ++j) b1[j] = load_b_frag(W, K, n0 + j * 16, 32);

    for (int k0 = 0; k0 < K; k0 += 64) {
#pragma unroll
        for (int j = 0; j < 4; ++j) acc[j] = wmma_f16(a0, b0[j], acc[j]);
        if (k0 + 64 < K) {
            __builtin_prefetch(W + (size_t)n0 * K + k0 + 128, 0, 3);
            a0 = load_a_frag(A, K, m0, k0 + 64);
#pragma unroll
            for (int j = 0; j < 4; ++j) b0[j] = load_b_frag(W, K, n0 + j * 16, k0 + 64);
        }
#pragma unroll
        for (int j = 0; j < 4; ++j) acc[j] = wmma_f16(a1, b1[j], acc[j]);
        if (k0 + 96 < K) {
            a1 = load_a_frag(A, K, m0, k0 + 96);
#pragma unroll
            for (int j = 0; j < 4; ++j) b1[j] = load_b_frag(W, K, n0 + j * 16, k0 + 96);
        }
    }
    // C/D layout: VGPR r -> row r (lanes 0-15) or r+8 (lanes 16-31), col = lane&15
    int lane = threadIdx.x & 31;
    int col = lane & 15, roff = (lane >> 4) << 3;
#pragma unroll
    for (int j = 0; j < 4; ++j) {
        int n = n0 + j * 16 + col;
        float bv = bias[n];
#pragma unroll
        for (int r = 0; r < 8; ++r)
            C[(size_t)(m0 + r + roff) * N + n] = acc[j][r] + bv;
    }
}

// ---------------- kernel 2: bias'd QKV -> RMSNorm(q,k) + RoPE + f16 repack ----------------
__global__ void norm_rope_kernel(const float* __restrict__ raw,   // [S][4608] (bias included)
                                 const float* __restrict__ gq, const float* __restrict__ gk,
                                 const float* __restrict__ fcos, const float* __restrict__ fsin,
                                 _Float16* __restrict__ qo,   // [NH][S][HD]
                                 _Float16* __restrict__ ko,   // [NH][S][HD]
                                 _Float16* __restrict__ vto)  // [NH][HD][S]
{
    int s = blockIdx.x, tid = threadIdx.x;
    const float* rq = raw + (size_t)s * QKVN;
    const float* rk = rq + DIM;
    const float* rv = rq + 2 * DIM;

    __shared__ float red[256];
    float pq = 0.f, pk = 0.f;
    for (int i = tid; i < DIM; i += 256) { float a = rq[i]; pq += a * a; float b = rk[i]; pk += b * b; }
    red[tid] = pq; __syncthreads();
    for (int st = 128; st > 0; st >>= 1) { if (tid < st) red[tid] += red[tid + st]; __syncthreads(); }
    float rsq_q = rsqrtf(red[0] / (float)DIM + 1e-6f);
    __syncthreads();
    red[tid] = pk; __syncthreads();
    for (int st = 128; st > 0; st >>= 1) { if (tid < st) red[tid] += red[tid + st]; __syncthreads(); }
    float rsq_k = rsqrtf(red[0] / (float)DIM + 1e-6f);

    // 3D grid position for RoPE bands (22 T / 21 H / 21 W complex dims per head)
    int fi = s / (H_G * W_G), rem = s % (H_G * W_G), hi = rem / W_G, wi = rem % W_G;

    for (int p = tid; p < DIM / 2; p += 256) {
        int d = 2 * p;
        int nh = d >> 7, dih = d & 127, c = dih >> 1;
        int pos = (c < 22) ? fi : ((c < 43) ? hi : wi);
        float cs = fcos[pos * CPLX + c], sn = fsin[pos * CPLX + c];
        size_t base = ((size_t)nh * S_LEN + s) * HD + dih;
        float a = rq[d] * rsq_q * gq[d];
        float b = rq[d + 1] * rsq_q * gq[d + 1];
        qo[base]     = (_Float16)(a * cs - b * sn);
        qo[base + 1] = (_Float16)(a * sn + b * cs);
        a = rk[d] * rsq_k * gk[d];
        b = rk[d + 1] * rsq_k * gk[d + 1];
        ko[base]     = (_Float16)(a * cs - b * sn);
        ko[base + 1] = (_Float16)(a * sn + b * cs);
    }
    for (int i = tid; i < DIM; i += 256) {
        int nh = i >> 7, dih = i & 127;
        vto[((size_t)nh * HD + dih) * S_LEN + s] = (_Float16)rv[i];
    }
}

// ---------------- kernel 3: flash attention (one wave = one head x 16 q rows) ----------------
__global__ void attn_kernel(const _Float16* __restrict__ q,  // [NH][S][HD]
                            const _Float16* __restrict__ k,  // [NH][S][HD]
                            const _Float16* __restrict__ vt, // [NH][HD][S]
                            _Float16* __restrict__ o)        // [S][DIM] (head nh at col nh*HD)
{
    __shared__ __align__(16) _Float16 pbuf[8][16 * 32];  // per-wave P staging tile
    int wave = (int)((blockIdx.x * blockDim.x + threadIdx.x) >> 5);
    const int QT = S_LEN / 16;
    if (wave >= NH * QT) return;
    int nh = wave % NH, qt = wave / NH;
    _Float16* pb = pbuf[threadIdx.x >> 5];
    int lane = threadIdx.x & 31;

    const _Float16* qh = q + (size_t)nh * S_LEN * HD;
    const _Float16* kh = k + (size_t)nh * S_LEN * HD;
    const _Float16* vh = vt + (size_t)nh * HD * S_LEN;
    int m0 = qt * 16;

    v16h qa[4];
#pragma unroll
    for (int d = 0; d < 4; ++d) qa[d] = load_a_frag(qh, HD, m0, d * 32);

    v8f oacc[8] = {};
    float rm[8], rl[8];
#pragma unroll
    for (int r = 0; r < 8; ++r) { rm[r] = -1e30f; rl[r] = 0.f; }
    const float scale = 0.08838834764831845f * 1.4426950408889634f;  // (1/sqrt(HD)) * log2(e)

    // prologue: K-fragments for the first key tile
    v16h kb[8];
#pragma unroll
    for (int d = 0; d < 4; ++d) {
        kb[d]     = load_b_frag(kh, HD, 0,  d * 32);
        kb[4 + d] = load_b_frag(kh, HD, 16, d * 32);
    }

    for (int j0 = 0; j0 < S_LEN; j0 += 32) {
        // ---- scores: 8 WMMAs (two interleaved accumulation chains) ----
        v8f s0 = {}, s1 = {};
#pragma unroll
        for (int d = 0; d < 4; ++d) {
            s0 = wmma_f16(qa[d], kb[d],     s0);
            s1 = wmma_f16(qa[d], kb[4 + d], s1);
        }
        // reload kb in place for the next key tile; these loads hide behind
        // the softmax VALU work, the LDS bounce and the PV WMMAs below.
        if (j0 + 32 < S_LEN) {
#pragma unroll
            for (int d = 0; d < 4; ++d) {
                kb[d]     = load_b_frag(kh, HD, j0 + 32, d * 32);
                kb[4 + d] = load_b_frag(kh, HD, j0 + 48, d * 32);
            }
        }
        float tm[8];
#pragma unroll
        for (int r = 0; r < 8; ++r) {
            s0[r] *= scale; s1[r] *= scale;
            tm[r] = fmaxf(s0[r], s1[r]);
        }
        // row-wise max across the 16 lanes of each half-wave
#pragma unroll
        for (int m = 1; m < 16; m <<= 1)
#pragma unroll
            for (int r = 0; r < 8; ++r) tm[r] = fmaxf(tm[r], __shfl_xor(tm[r], m, 32));
        // online softmax update (log2 domain; scale already folds in log2(e))
        float fct[8], ts[8];
#pragma unroll
        for (int r = 0; r < 8; ++r) {
            float mn = fmaxf(rm[r], tm[r]);
            fct[r] = exp2f(rm[r] - mn);
            rm[r] = mn;
            s0[r] = exp2f(s0[r] - mn);
            s1[r] = exp2f(s1[r] - mn);
            ts[r] = s0[r] + s1[r];
        }
#pragma unroll
        for (int m = 1; m < 16; m <<= 1)
#pragma unroll
            for (int r = 0; r < 8; ++r) ts[r] += __shfl_xor(ts[r], m, 32);
#pragma unroll
        for (int r = 0; r < 8; ++r) rl[r] = rl[r] * fct[r] + ts[r];
#pragma unroll
        for (int t = 0; t < 8; ++t)
#pragma unroll
            for (int r = 0; r < 8; ++r) oacc[t][r] *= fct[r];

        // ---- C-layout -> A-layout via LDS bounce (f16 P tile, row-major 16x32) ----
        int col = lane & 15, roff = (lane >> 4) << 3;
#pragma unroll
        for (int r = 0; r < 8; ++r) {
            pb[(r + roff) * 32 + col]      = (_Float16)s0[r];
            pb[(r + roff) * 32 + 16 + col] = (_Float16)s1[r];
        }
        __asm__ volatile("s_wait_dscnt 0" ::: "memory");
        v16h pa = load_a_frag(pb, 32, 0, 0);
        __asm__ volatile("" ::: "memory");

        // ---- P @ V : batch all 8 V-fragments, then 8 independent WMMA chains ----
        v16h vb[8];
#pragma unroll
        for (int t = 0; t < 8; ++t) vb[t] = load_b_frag(vh, S_LEN, t * 16, j0);
#pragma unroll
        for (int t = 0; t < 8; ++t) oacc[t] = wmma_f16(pa, vb[t], oacc[t]);
    }

    // epilogue: normalize and write [S][DIM] f16 for the output projection
    int col = lane & 15, roff = (lane >> 4) << 3;
#pragma unroll
    for (int t = 0; t < 8; ++t) {
#pragma unroll
        for (int r = 0; r < 8; ++r) {
            o[(size_t)(m0 + r + roff) * DIM + nh * HD + t * 16 + col] =
                (_Float16)(oacc[t][r] / rl[r]);
        }
    }
}

// ---------------- host launch ----------------
extern "C" void kernel_launch(void* const* d_in, const int* in_sizes, int n_in,
                              void* d_out, int out_size, void* d_ws, size_t ws_size,
                              hipStream_t stream) {
    const float* x   = (const float*)d_in[0];
    const float* Wq  = (const float*)d_in[1];
    const float* bq  = (const float*)d_in[2];
    const float* Wk  = (const float*)d_in[3];
    const float* bk  = (const float*)d_in[4];
    const float* Wv  = (const float*)d_in[5];
    const float* bv  = (const float*)d_in[6];
    const float* Wo  = (const float*)d_in[7];
    const float* bo  = (const float*)d_in[8];
    const float* gq  = (const float*)d_in[9];
    const float* gk  = (const float*)d_in[10];
    const float* fc  = (const float*)d_in[11];
    const float* fs  = (const float*)d_in[12];
    float* out = (float*)d_out;

    // workspace layout (bytes, all 256B-aligned)
    char* ws = (char*)d_ws;
    size_t off = 0;
    _Float16* xh    = (_Float16*)(ws + off); off += (size_t)S_LEN * DIM * 2;   // x f16
    _Float16* wqkvh = (_Float16*)(ws + off); off += (size_t)QKVN * DIM * 2;    // packed QKV weight f16
    _Float16* woh   = (_Float16*)(ws + off); off += (size_t)DIM * DIM * 2;     // Wo f16
    float*    bqkv  = (float*)   (ws + off); off += (size_t)QKVN * 4;          // packed bias f32
    size_t rawOff = off;
    float*    qkvraw= (float*)   (ws + off); off += (size_t)S_LEN * QKVN * 4;  // qkv raw f32
    _Float16* qh    = (_Float16*)(ws + off); off += (size_t)S_LEN * DIM * 2;   // q f16 [NH][S][HD]
    _Float16* kh    = (_Float16*)(ws + off); off += (size_t)S_LEN * DIM * 2;   // k f16 [NH][S][HD]
    _Float16* vth   = (_Float16*)(ws + off); off += (size_t)S_LEN * DIM * 2;   // v^T f16 [NH][HD][S]
    _Float16* oh    = (_Float16*)(ws + rawOff);                                // reuse raw region

    // 0) convert/pack
    prep_kernel<<<4096, 256, 0, stream>>>(x, Wq, Wk, Wv, Wo, bq, bk, bv, xh, wqkvh, woh, bqkv);

    // 1) fused QKV projection: [3840x1536] @ [4608x1536]^T  (240 x 72 wave-tiles)
    {
        int mtiles = S_LEN / 16, ntiles = QKVN / 64;
        int waves = mtiles * ntiles;
        gemm_wmma<<<(waves + 7) / 8, 256, 0, stream>>>(xh, wqkvh, bqkv, qkvraw,
                                                       S_LEN, QKVN, DIM, mtiles, ntiles);
    }

    // 2) RMSNorm + RoPE + f16 repack
    norm_rope_kernel<<<S_LEN, 256, 0, stream>>>(qkvraw, gq, gk, fc, fs, qh, kh, vth);

    // 3) flash attention (2880 waves)
    attn_kernel<<<(NH * (S_LEN / 16) + 7) / 8, 256, 0, stream>>>(qh, kh, vth, oh);

    // 4) output projection: [3840x1536] @ [1536x1536]^T + bo -> d_out f32
    {
        int mtiles = S_LEN / 16, ntiles = DIM / 64;
        int waves = mtiles * ntiles;
        gemm_wmma<<<(waves + 7) / 8, 256, 0, stream>>>(oh, woh, bo, out,
                                                       S_LEN, DIM, DIM, mtiles, ntiles);
    }
}
